// NeuraLogic_57174604644834
// MI455X (gfx1250) — matrix operand order repaired and checked by reading, not compiled, not run on hardware
//
#include <hip/hip_runtime.h>
#include <hip/hip_bf16.h>

#define D64 64

typedef __attribute__((ext_vector_type(16))) __bf16 v16bf;
typedef __attribute__((ext_vector_type(8)))  float  v8f;

// ---------------- weight pre-pack: f32 [64,64] -> bf16 in WMMA B-frag order --
// Fragment f = nt*2 + kc (nt = 16-col tile 0..3, kc = K-chunk 0..1).
// Within a fragment: lane L holds col N = nt*16 + L%16,
// elements e=0..15 cover K = kc*32 + (L>=16 ? 16 : 0) + e.
// Packed index = ((f*32 + lane)*16 + e)  -> each lane's 16 bf16 are contiguous.
__global__ void pack_w_bf16(const float* __restrict__ W, __bf16* __restrict__ P) {
  int i = blockIdx.x * blockDim.x + threadIdx.x;
  if (i >= 4096) return;
  int e    = i & 15;
  int lane = (i >> 4) & 31;
  int f    = i >> 9;           // 0..7
  int nt   = f >> 1, kc = f & 1;
  int k = kc * 32 + ((lane & 16) ? 16 : 0) + e;
  int n = nt * 16 + (lane & 15);
  P[i] = (__bf16)W[k * D64 + n];
}

// ---------------- A fragment builder (16-bit A 16x32, k = kb + 16*(lane/16) + e, same as packed B) -------
__device__ __forceinline__ v16bf make_a_frag(const float* __restrict__ A,
                                             int row, int kb, int lane) {
  const float* r = A + (size_t)row * D64 + kb + ((lane & 16) ? 16 : 0);
  v16bf a;
#pragma unroll
  for (int e = 0; e < 8; ++e) a[e] = (__bf16)r[e];
#pragma unroll
  for (int e = 0; e < 8; ++e) a[8 + e] = (__bf16)r[8 + e];
  return a;
}

// C = act( sum_p A_p @ W_p [+ bias] ).  A_p:[nrows,64] f32, W_p: packed bf16.
// B fragments + bias hoisted into registers; persistent grid-stride tile loop.
template <int NP>
__global__ __launch_bounds__(256) void gemm64_fused(
    const float* A0, const __bf16* W0,
    const float* A1, const __bf16* W1,
    const float* A2, const __bf16* W2,
    const float* bias, float* __restrict__ C,
    int nrows, int do_relu) {
  const int lane = threadIdx.x & 31;
  const int wid  = (int)((blockIdx.x * blockDim.x + threadIdx.x) >> 5);
  const int nw   = (int)((gridDim.x * blockDim.x) >> 5);
  const int ntiles = nrows >> 4;  // all row counts are multiples of 16
  const int nloc = lane & 15;

  const float*  As[3] = {A0, A1, A2};
  const __bf16* Ws[3] = {W0, W1, W2};

  // Hoisted weight fragments: coalesced 2x b128 per fragment, no conversions.
  v16bf bfr[NP][8];
#pragma unroll
  for (int p = 0; p < NP; ++p)
#pragma unroll
    for (int f = 0; f < 8; ++f)
      bfr[p][f] = *(const v16bf*)(Ws[p] + ((size_t)(f * 32 + lane) * 16));

  // Hoisted bias: col = t*16 + lane%16 is tile-invariant.
  float bv[4];
#pragma unroll
  for (int t = 0; t < 4; ++t) bv[t] = bias ? bias[t * 16 + nloc] : 0.f;

  for (int tile = wid; tile < ntiles; tile += nw) {
    const int arow = (tile << 4) + nloc;
    v8f zero8 = {0.f, 0.f, 0.f, 0.f, 0.f, 0.f, 0.f, 0.f};
    v8f acc[4] = {zero8, zero8, zero8, zero8};

#pragma unroll
    for (int p = 0; p < NP; ++p) {
      v16bf a0 = make_a_frag(As[p], arow, 0, lane);
      v16bf a1 = make_a_frag(As[p], arow, 32, lane);
#pragma unroll
      for (int t = 0; t < 4; ++t) {
        acc[t] = __builtin_amdgcn_wmma_f32_16x16x32_bf16(
            false, a0, false, bfr[p][t * 2 + 0], (short)0, acc[t], false, false);
        acc[t] = __builtin_amdgcn_wmma_f32_16x16x32_bf16(
            false, a1, false, bfr[p][t * 2 + 1], (short)0, acc[t], false, false);
      }
    }

    // D layout: elem r -> row = r + (lane>=16 ? 8 : 0), col = t*16 + lane%16
    const int mbase = (tile << 4) + ((lane & 16) ? 8 : 0);
#pragma unroll
    for (int t = 0; t < 4; ++t) {
      const int col = t * 16 + nloc;
#pragma unroll
      for (int r = 0; r < 8; ++r) {
        float v = acc[t][r] + bv[t];
        if (do_relu) v = v > 0.f ? v : 0.f;
        C[(size_t)(mbase + r) * D64 + col] = v;
      }
    }
  }
}

// ---------------- elementwise / scatter kernels -----------------------------

__global__ void zero_f32(float* __restrict__ p, size_t n) {
  size_t i = (size_t)blockIdx.x * blockDim.x + threadIdx.x;
  size_t stride = (size_t)gridDim.x * blockDim.x;
  for (; i < n; i += stride) p[i] = 0.f;
}

__global__ void relu_f32(float* __restrict__ p, size_t n) {
  size_t i = (size_t)blockIdx.x * blockDim.x + threadIdx.x;
  if (i < n) { float v = p[i]; p[i] = v > 0.f ? v : 0.f; }
}

__global__ void add_f32(const float* __restrict__ a, const float* __restrict__ b,
                        float* __restrict__ c, size_t n) {
  size_t i = (size_t)blockIdx.x * blockDim.x + threadIdx.x;
  if (i < n) c[i] = a[i] + b[i];
}

// h[r*64+d] /= max(den[r % den_count], 1)
__global__ void rowdiv_f32(float* __restrict__ h, const float* __restrict__ den,
                           int nrows, int den_count) {
  size_t i = (size_t)blockIdx.x * blockDim.x + threadIdx.x;
  if (i >= (size_t)nrows * D64) return;
  int r = (int)(i >> 6);
  float d = den[r % den_count];
  h[i] /= (d > 1.f ? d : 1.f);
}

// out[idx[t]] += 1  (degree / graph-size counting)
__global__ void count_scatter(const int* __restrict__ idx, float* __restrict__ out,
                              int n) {
  int t = blockIdx.x * blockDim.x + threadIdx.x;
  if (t < n) atomicAdd(out + idx[t], 1.f);
}

// out[dst[e]] += feat[src[e]]   (16 threads/edge, 4 floats each)
__global__ void scatter_rows(const float* __restrict__ feat,
                             const int* __restrict__ src,
                             const int* __restrict__ dst,
                             float* __restrict__ out, int E) {
  int t = blockIdx.x * blockDim.x + threadIdx.x;
  int e = t >> 4;
  if (e >= E) return;
  int c = (t & 15) * 4;
  int u = src[e], v = dst[e];
  const float4 val = *(const float4*)(feat + (size_t)u * D64 + c);
  float* o = out + (size_t)v * D64 + c;
  atomicAdd(o + 0, val.x); atomicAdd(o + 1, val.y);
  atomicAdd(o + 2, val.z); atomicAdd(o + 3, val.w);
}

// GCN layer folded: h1acc[dst[e]] += embW[x[src[e]]]  (embW = emb @ W_gcn, L2-resident)
__global__ void gcn_scatter(const float* __restrict__ embW,
                            const int* __restrict__ x,
                            const int* __restrict__ src,
                            const int* __restrict__ dst,
                            float* __restrict__ out, int E) {
  int t = blockIdx.x * blockDim.x + threadIdx.x;
  int e = t >> 4;
  if (e >= E) return;
  int c = (t & 15) * 4;
  int u = src[e], v = dst[e];
  int xi = x[u];
  const float4 val = *(const float4*)(embW + (size_t)xi * D64 + c);
  float* o = out + (size_t)v * D64 + c;
  atomicAdd(o + 0, val.x); atomicAdd(o + 1, val.y);
  atomicAdd(o + 2, val.z); atomicAdd(o + 3, val.w);
}

// g_k[batch[n]] += h_k[n] for k=1..3
__global__ void pool_scatter(const float* __restrict__ h1, const float* __restrict__ h2,
                             const float* __restrict__ h3, const int* __restrict__ batch,
                             float* __restrict__ g1, float* __restrict__ g2,
                             float* __restrict__ g3, int N) {
  int t = blockIdx.x * blockDim.x + threadIdx.x;
  int n = t >> 4;
  if (n >= N) return;
  int c = (t & 15) * 4;
  int b = batch[n];
  size_t si = (size_t)n * D64 + c, di = (size_t)b * D64 + c;
  float4 v1 = *(const float4*)(h1 + si);
  float4 v2 = *(const float4*)(h2 + si);
  float4 v3 = *(const float4*)(h3 + si);
  atomicAdd(g1 + di + 0, v1.x); atomicAdd(g1 + di + 1, v1.y);
  atomicAdd(g1 + di + 2, v1.z); atomicAdd(g1 + di + 3, v1.w);
  atomicAdd(g2 + di + 0, v2.x); atomicAdd(g2 + di + 1, v2.y);
  atomicAdd(g2 + di + 2, v2.z); atomicAdd(g2 + di + 3, v2.w);
  atomicAdd(g3 + di + 0, v3.x); atomicAdd(g3 + di + 1, v3.y);
  atomicAdd(g3 + di + 2, v3.z); atomicAdd(g3 + di + 3, v3.w);
}

// ---------------- launcher ---------------------------------------------------

static inline int nblk(size_t n, int t) { return (int)((n + t - 1) / t); }
static inline int gemm_grid(int rows) {
  int b = ((rows >> 4) + 7) / 8;           // one 16-row tile per wave, 8 waves/block
  return b < 128 ? (b < 1 ? 1 : b) : 128;  // cap -> persistent waves reuse B frags
}

extern "C" void kernel_launch(void* const* d_in, const int* in_sizes, int n_in,
                              void* d_out, int out_size, void* d_ws, size_t ws_size,
                              hipStream_t stream) {
  const int*   x        = (const int*)d_in[0];
  const int*   ei       = (const int*)d_in[1];
  const int*   batch    = (const int*)d_in[2];
  const float* emb      = (const float*)d_in[3];
  const float* W_gcn    = (const float*)d_in[4];
  const float* W_sage_l = (const float*)d_in[5];
  const float* W_sage_r = (const float*)d_in[6];
  const float* W_gin1   = (const float*)d_in[7];
  const float* b_gin1   = (const float*)d_in[8];
  const float* W_gin2   = (const float*)d_in[9];
  const float* b_gin2   = (const float*)d_in[10];
  const float* W_pool1  = (const float*)d_in[11];
  const float* W_pool2  = (const float*)d_in[12];
  const float* W_pool3  = (const float*)d_in[13];

  const int N = in_sizes[0];
  const int E = in_sizes[1] / 2;
  const int V = in_sizes[3] / D64;
  const int G = out_size / D64;
  const int* src = ei;       // edge_index[0]
  const int* dst = ei + E;   // edge_index[1]

  float* ws = (float*)d_ws;
  size_t off = 0;
  auto take = [&](size_t n) { float* p = ws + off; off += n; return p; };
  float* embW = take((size_t)V * D64);
  float* h1   = take((size_t)N * D64);
  float* h2   = take((size_t)N * D64);
  float* h3   = take((size_t)N * D64);
  float* t0   = take((size_t)N * D64);
  float* t1   = take((size_t)N * D64);
  float* deg  = take((size_t)N);
  float* cnt  = take((size_t)G);
  float* g1   = take((size_t)G * D64);
  float* g2   = take((size_t)G * D64);
  float* g3   = take((size_t)G * D64);
  // 8 packed bf16 weight matrices (4096 bf16 = 2048 floats each, 256B aligned)
  __bf16* pW[8];
  for (int i = 0; i < 8; ++i) pW[i] = (__bf16*)take(2048);
  (void)ws_size; (void)n_in;

  const int T = 256;
  const size_t ND = (size_t)N * D64;

  // Pre-pack all weight matrices into WMMA B-fragment order (bf16).
  const float* rawW[8] = {W_gcn, W_sage_l, W_sage_r, W_gin1,
                          W_gin2, W_pool1, W_pool2, W_pool3};
  for (int i = 0; i < 8; ++i)
    pack_w_bf16<<<16, T, 0, stream>>>(rawW[i], pW[i]);

  // embW = emb @ W_gcn   (folds GCN's node-level GEMM into a 128x64 table)
  gemm64_fused<1><<<gemm_grid(V), T, 0, stream>>>(
      emb, pW[0], nullptr, nullptr, nullptr, nullptr, nullptr, embW, V, 0);

  // h1 = relu( scatter_sum_{e: src->dst} embW[x[src]] )
  zero_f32<<<nblk(ND, T), T, 0, stream>>>(h1, ND);
  gcn_scatter<<<nblk((size_t)E * 16, T), T, 0, stream>>>(embW, x, src, dst, h1, E);
  relu_f32<<<nblk(ND, T), T, 0, stream>>>(h1, ND);

  // deg + mean aggregation of h1
  zero_f32<<<nblk((size_t)N, T), T, 0, stream>>>(deg, (size_t)N);
  count_scatter<<<nblk((size_t)E, T), T, 0, stream>>>(dst, deg, E);
  zero_f32<<<nblk(ND, T), T, 0, stream>>>(t0, ND);
  scatter_rows<<<nblk((size_t)E * 16, T), T, 0, stream>>>(h1, src, dst, t0, E);
  rowdiv_f32<<<nblk(ND, T), T, 0, stream>>>(t0, deg, N, N);

  // h2 = relu(mean_nb @ W_sage_l + h1 @ W_sage_r)
  gemm64_fused<2><<<gemm_grid(N), T, 0, stream>>>(
      t0, pW[1], h1, pW[2], nullptr, nullptr, nullptr, h2, N, 1);

  // GIN: z = h2 + scatter_sum(h2); h3 = relu(relu(z@W1+b1)@W2+b2)
  zero_f32<<<nblk(ND, T), T, 0, stream>>>(t1, ND);
  scatter_rows<<<nblk((size_t)E * 16, T), T, 0, stream>>>(h2, src, dst, t1, E);
  add_f32<<<nblk(ND, T), T, 0, stream>>>(h2, t1, t0, ND);
  gemm64_fused<1><<<gemm_grid(N), T, 0, stream>>>(
      t0, pW[3], nullptr, nullptr, nullptr, nullptr, b_gin1, t1, N, 1);
  gemm64_fused<1><<<gemm_grid(N), T, 0, stream>>>(
      t1, pW[4], nullptr, nullptr, nullptr, nullptr, b_gin2, h3, N, 1);

  // Global mean pooling + jumping-knowledge projection
  zero_f32<<<nblk((size_t)G, T), T, 0, stream>>>(cnt, (size_t)G);
  count_scatter<<<nblk((size_t)N, T), T, 0, stream>>>(batch, cnt, N);
  zero_f32<<<nblk((size_t)3 * G * D64, T), T, 0, stream>>>(g1, (size_t)3 * G * D64);
  pool_scatter<<<nblk((size_t)N * 16, T), T, 0, stream>>>(h1, h2, h3, batch, g1, g2, g3, N);
  rowdiv_f32<<<nblk((size_t)3 * G * D64, T), T, 0, stream>>>(g1, cnt, 3 * G, G);

  // out = relu(g1@Wp1 + g2@Wp2 + g3@Wp3)
  gemm64_fused<3><<<gemm_grid(G), T, 0, stream>>>(
      g1, pW[5], g2, pW[6], g3, pW[7], nullptr, (float*)d_out, G, 1);
}
